// MoEBlock_35107062678429
// MI455X (gfx1250) — matrix operand order repaired and checked by reading, not compiled, not run on hardware
//
#include <hip/hip_runtime.h>

#define HIDDEN 1024
#define INTER  704
#define NEXP   32
#define NTOK   8192
#define MAXTILES ((NTOK / 16) + NEXP)   // 544 upper bound on ragged 16-row tiles

typedef __attribute__((ext_vector_type(16))) __bf16         v16bf;
typedef __attribute__((ext_vector_type(16))) unsigned short v16u;
typedef __attribute__((ext_vector_type(8)))  float          v8f;

// Native fp32 -> bf16 convert (RNE); backend emits v_cvt_pk_bf16_f32 pairs.
__device__ __forceinline__ unsigned short bf16_bits(float f) {
  return __builtin_bit_cast(unsigned short, (__bf16)f);
}

__device__ __forceinline__ unsigned pack_bf16x2(float a, float b) {
  return (unsigned)bf16_bits(a) | ((unsigned)bf16_bits(b) << 16);
}

// Low 32 bits of a flat shared-memory address are the LDS byte offset
// (ISA: LDS aperture access uses addr[31:0]).
__device__ __forceinline__ unsigned lds_off32(const void* p) {
  return (unsigned)(unsigned long long)(uintptr_t)p;
}

// Map global tile index -> expert id, row range [rs, re)
__device__ __forceinline__ void tile_lookup(const int* __restrict__ tpe, int t,
                                            int* s_info) {
  int rs = 0, re = 0, e = -1, acc = 0;
  for (int i = 0; i < NEXP; ++i) {
    int cnt = tpe[i];
    int tiles = (cnt + 15) >> 4;
    if (t < tiles) {
      e = i;
      rs = acc + t * 16;
      re = acc + cnt;
      if (re > rs + 16) re = rs + 16;
      break;
    }
    t -= tiles;
    acc += cnt;
  }
  s_info[0] = e; s_info[1] = rs; s_info[2] = re;
}

// A-fragment LDS slot for element (m, k) of a 16x32 bf16 tile:
// lane = m + 16*((k>>4)&1), element = k&15 (contiguous k = 16h + s).
// For k aligned to 4 the quad is contiguous -> single 8-byte store/copy.

// ---------------------------------------------------------------------------
// Kernel 1: gate_up = x @ w1w3[e]; h = silu(gate)*up stored bf16.
// grid = (MAXTILES, INTER/64), block = 128 (4 waves).
// Each wave owns one 16x16 h tile and computes both its gate and up tiles
// (2 WMMAs / K-chunk, shared A fragment); SwiGLU fused in-register.
// Ragged guard: staging rows are CLAMPED (garbage rows only affect output
// rows that are never stored), so no masking is needed.
// ---------------------------------------------------------------------------
__global__ void __launch_bounds__(128)
moe_gemm1_swiglu(const float* __restrict__ x,
                 const int*   __restrict__ tpe,
                 const float* __restrict__ w1w3,
                 unsigned short* __restrict__ hbuf) {
  __shared__ int s_info[3];
  __shared__ __attribute__((aligned(32))) unsigned short lds_a[32 * 16];

  if (threadIdx.x == 0) tile_lookup(tpe, (int)blockIdx.x, s_info);
  __syncthreads();
  const int expert = s_info[0];
  if (expert < 0) return;
  const int rs = s_info[1], re = s_info[2];

  const int wave = threadIdx.x >> 5;
  const int lane = threadIdx.x & 31;

  // A-staging: thread t -> row m = t>>3, k-quad = (t&7)*4 (clamped row)
  const int sm = threadIdx.x >> 3;
  const int sk = (threadIdx.x & 7) << 2;
  const int srow = (rs + sm < re) ? (rs + sm) : rs;
  const int slot = (sm + ((sk & 16) ? 16 : 0)) * 16 + (sk & 15);

  const int hcol = blockIdx.y * 64 + (wave << 4) + (lane & 15);
  const int kb = (lane < 16) ? 0 : 16;             // B-fragment K base
  const float* __restrict__ wp = w1w3 + (size_t)expert * HIDDEN * (2 * INTER);

  v8f accG = {}, accU = {};
  for (int kc = 0; kc < HIDDEN; kc += 32) {
    __syncthreads();
    {
      float4 v = *(const float4*)(x + (size_t)srow * HIDDEN + kc + sk);
      uint2 p; p.x = pack_bf16x2(v.x, v.y); p.y = pack_bf16x2(v.z, v.w);
      *(uint2*)(&lds_a[slot]) = p;
    }
    __syncthreads();

    v16u au = *(const v16u*)(&lds_a[lane << 4]);
    v16bf a = __builtin_bit_cast(v16bf, au);

    const float* wk = wp + (size_t)(kc + kb) * (2 * INTER) + hcol;
    if (kc + 32 < HIDDEN)
      __builtin_prefetch(wk + (size_t)32 * (2 * INTER), 0, 1);
    v16bf bg, bu;
#pragma unroll
    for (int i = 0; i < 16; ++i) {
      bg[i] = (__bf16)wk[(size_t)i * (2 * INTER)];            // gate col
      bu[i] = (__bf16)wk[(size_t)i * (2 * INTER) + INTER];    // up col
    }
    accG = __builtin_amdgcn_wmma_f32_16x16x32_bf16(
        false, a, false, bg, (short)0, accG, false, false);
    accU = __builtin_amdgcn_wmma_f32_16x16x32_bf16(
        false, a, false, bu, (short)0, accU, false, false);
  }

  // SwiGLU in-register: gate/up accumulators are lane/element aligned.
  const int mbase = (lane < 16) ? 0 : 8;
#pragma unroll
  for (int r = 0; r < 8; ++r) {
    float g = accG[r];
    float s = g / (1.0f + __expf(-g));   // silu
    float hv = s * accU[r];
    int row = rs + r + mbase;            // C/D layout: VGPR r -> row
    if (row < re)
      hbuf[(size_t)row * INTER + hcol] = bf16_bits(hv);
  }
}

// ---------------------------------------------------------------------------
// Kernel 2: out = h @ w2[e].
// grid = (MAXTILES, HIDDEN/128), block = 128 (4 waves x 2 N-tiles = 128 cols).
// A tile is already bf16 -> staged with GLOBAL_LOAD_ASYNC_TO_LDS_B64:
// each lane copies its contiguous 4-half quad straight into the
// fragment-order LDS slot (ASYNCcnt path, no VGPR round-trip).
// ---------------------------------------------------------------------------
__global__ void __launch_bounds__(128)
moe_gemm2(const unsigned short* __restrict__ hbuf,
          const int*   __restrict__ tpe,
          const float* __restrict__ w2,
          float* __restrict__ out) {
  __shared__ int s_info[3];
  __shared__ __attribute__((aligned(32))) unsigned short lds_a[32 * 16];

  if (threadIdx.x == 0) tile_lookup(tpe, (int)blockIdx.x, s_info);
  __syncthreads();
  const int expert = s_info[0];
  if (expert < 0) return;
  const int rs = s_info[1], re = s_info[2];

  const int wave = threadIdx.x >> 5;
  const int lane = threadIdx.x & 31;

  const int sm = threadIdx.x >> 3;
  const int sk = (threadIdx.x & 7) << 2;
  const int srow = (rs + sm < re) ? (rs + sm) : rs;
  const int slot = (sm + ((sk & 16) ? 16 : 0)) * 16 + (sk & 15);
  const unsigned ldsdst = lds_off32(&lds_a[slot]);

  const int col0 = blockIdx.y * 128 + (wave << 5) + (lane & 15);
  const int kb = (lane < 16) ? 0 : 16;
  const float* __restrict__ wp = w2 + (size_t)expert * INTER * HIDDEN;

  v8f acc0 = {}, acc1 = {};
  for (int kc = 0; kc < INTER; kc += 32) {
    __syncthreads();
    {
      const unsigned short* src = hbuf + (size_t)srow * INTER + kc + sk;
      // async DMA: 8 bytes global -> fragment-order LDS slot, per lane
      asm volatile("global_load_async_to_lds_b64 %0, %1, off"
                   :: "v"(ldsdst), "v"(src) : "memory");
      asm volatile("s_wait_asynccnt 0x0" ::: "memory");
    }
    __syncthreads();

    v16u au = *(const v16u*)(&lds_a[lane << 4]);
    v16bf a = __builtin_bit_cast(v16bf, au);

    const float* wk = wp + (size_t)(kc + kb) * HIDDEN + col0;
    if (kc + 32 < INTER)
      __builtin_prefetch(wk + (size_t)32 * HIDDEN, 0, 1);
    v16bf b0, b1;
#pragma unroll
    for (int i = 0; i < 16; ++i) {
      b0[i] = (__bf16)wk[(size_t)i * HIDDEN];
      b1[i] = (__bf16)wk[(size_t)i * HIDDEN + 16];
    }
    acc0 = __builtin_amdgcn_wmma_f32_16x16x32_bf16(
        false, a, false, b0, (short)0, acc0, false, false);
    acc1 = __builtin_amdgcn_wmma_f32_16x16x32_bf16(
        false, a, false, b1, (short)0, acc1, false, false);
  }

  const int mbase = (lane < 16) ? 0 : 8;
#pragma unroll
  for (int r = 0; r < 8; ++r) {
    int row = rs + r + mbase;
    if (row < re) {
      out[(size_t)row * HIDDEN + col0]      = acc0[r];
      out[(size_t)row * HIDDEN + col0 + 16] = acc1[r];
    }
  }
}

extern "C" void kernel_launch(void* const* d_in, const int* in_sizes, int n_in,
                              void* d_out, int out_size, void* d_ws, size_t ws_size,
                              hipStream_t stream) {
  const float* x    = (const float*)d_in[0];
  const int*   tpe  = (const int*)d_in[1];
  const float* w1w3 = (const float*)d_in[2];
  const float* w2   = (const float*)d_in[3];
  float*       out  = (float*)d_out;
  unsigned short* hbuf = (unsigned short*)d_ws;  // bf16 h: [8192,704] = 11.5 MB

  dim3 block(128);
  moe_gemm1_swiglu<<<dim3(MAXTILES, INTER / 64), block, 0, stream>>>(
      x, tpe, w1w3, hbuf);
  moe_gemm2<<<dim3(MAXTILES, HIDDEN / 128), block, 0, stream>>>(
      hbuf, tpe, w2, out);
}
